// H2ONet_Decoder_69630009803201
// MI455X (gfx1250) — matrix operand order repaired and checked by reading, not compiled, not run on hardware
//
#include <hip/hip_runtime.h>

// ---------------------------------------------------------------------------
// H2ONet decoder for MI455X (gfx1250): bf16 WMMA GEMMs (f32 accumulate),
// double-buffered async global->LDS tile staging (ASYNCcnt), fused
// spiral-gather GEMM. All activation matrices are column-major:
//   out[n * M + m]  == B-matrix of the next GEMM, already k-contiguous per col.
// ---------------------------------------------------------------------------

typedef unsigned short u16;
typedef unsigned int   u32;
typedef unsigned long long u64;

typedef __attribute__((ext_vector_type(16))) __bf16 v16bf;
typedef __attribute__((ext_vector_type(8)))  float  v8f;

union Frag16 { v16bf v; u32 u[8]; };

__device__ __forceinline__ u16 f2bf(float f) {
    union { float f; u32 u; } x; x.f = f;
    u32 r = x.u + 0x7FFFu + ((x.u >> 16) & 1u);   // round-to-nearest-even
    return (u16)(r >> 16);
}
__device__ __forceinline__ float bf2f(u16 h) {
    union { u32 u; float f; } x; x.u = ((u32)h) << 16;
    return x.f;
}

// LDS byte offset of a __shared__ pointer (flat LDS aperture: low 32 bits)
__device__ __forceinline__ u32 lds_off(const void* p) {
    return (u32)(uintptr_t)p;
}
// Per-lane async copy of 16 bytes global -> LDS (tracked by ASYNCcnt)
__device__ __forceinline__ void async_ld_b128(u32 dst_lds, const void* src) {
    asm volatile("global_load_async_to_lds_b128 %0, %1, off"
                 :: "v"(dst_lds), "v"((u64)(uintptr_t)src) : "memory");
}
__device__ __forceinline__ void wait_async0() {
    asm volatile("s_wait_asynccnt 0x0" ::: "memory");
}

// ---------------------------------------------------------------------------
// Tiled bf16 WMMA GEMM:  out[n*M+m] = relu(sum_k A[m*K+k] * B[n*K+k] + bias[m])
// A row-major [MxK]; B column-major [KxN] (ld = K); out column-major.
// Block tile 64(M) x 128(N), 8 waves = 4(M) x 2(N); wave = 1x4 WMMA tiles.
// Double-buffered LDS; async tile i+1 overlaps WMMA on tile i.
// Requires M%64==0, N%128==0, K%32==0 (true at every call site).
// ---------------------------------------------------------------------------
__global__ __launch_bounds__(256)
void k_gemm(const u16* __restrict__ A, const u16* __restrict__ Bm,
            u16* __restrict__ out, const float* __restrict__ bias,
            int M, int N, int K, int relu)
{
    __shared__ u16 As[2][64][40];    // [buf][m][k], row stride 80B (16B aligned)
    __shared__ u16 Bs[2][128][40];   // [buf][n][k], k-contiguous

    const int t    = threadIdx.x;
    const int m0   = blockIdx.y * 64;
    const int n0   = blockIdx.x * 128;
    const int wave = t >> 5, lane = t & 31;
    const int lq   = lane & 15, half = lane >> 4;
    const int mw   = (wave >> 1) << 4;    // 0,16,32,48
    const int nw   = (wave & 1) << 6;     // 0,64

    v8f acc[4] = {v8f{}, v8f{}, v8f{}, v8f{}};

    const int a_m = t >> 2, a_k = (t & 3) << 3;   // A: 64 rows x 4 chunks of 8
    const int b_n = t >> 2, b_k = (t & 3) << 3;   // B: 128 cols (2 halves) x 4 chunks

    const u32 dA[2]  = {lds_off(&As[0][a_m][a_k]),      lds_off(&As[1][a_m][a_k])};
    const u32 dB0[2] = {lds_off(&Bs[0][b_n][b_k]),      lds_off(&Bs[1][b_n][b_k])};
    const u32 dB1[2] = {lds_off(&Bs[0][b_n + 64][b_k]), lds_off(&Bs[1][b_n + 64][b_k])};

    const u16* pA  = A  + (size_t)(m0 + a_m) * K + a_k;
    const u16* pB0 = Bm + (size_t)(n0 + b_n) * K + b_k;
    const u16* pB1 = Bm + (size_t)(n0 + b_n + 64) * K + b_k;

    // prologue: tile 0 -> buffer 0
    async_ld_b128(dA[0],  pA);
    async_ld_b128(dB0[0], pB0);
    async_ld_b128(dB1[0], pB1);

    const int nk = K >> 5;
    for (int i = 0; i < nk; ++i) {
        const int cur = i & 1, nxt = cur ^ 1;

        wait_async0();        // this wave's tile i has landed in LDS
        __syncthreads();      // all waves landed; all reads of buf[nxt] (iter i-1) done

        if (i + 1 < nk) {     // prefetch tile i+1, overlapping the WMMAs below
            const int k1 = (i + 1) << 5;
            async_ld_b128(dA[nxt],  pA  + k1);
            async_ld_b128(dB0[nxt], pB0 + k1);
            async_ld_b128(dB1[nxt], pB1 + k1);
        }

        Frag16 af;
        #pragma unroll
        for (int j = 0; j < 8; ++j) {
            // ISA 16-bit A 16x32 fragment: VGPR j<4 -> K=2j(+8*half), j>=4 -> 16+2(j-4)(+8*half)
            int ka = ((j & 4) << 2) + ((j & 3) << 1) + (half << 3);
            af.u[j] = *(const u32*)&As[cur][mw + lq][ka];
        }
        #pragma unroll
        for (int nt = 0; nt < 4; ++nt) {
            Frag16 bf;
            #pragma unroll
            for (int j = 0; j < 8; ++j) {
                // 16-bit B KxN fragment, same k order as A: VGPR j<4 -> K=2j(+8*half), j>=4 -> 16+2(j-4)(+8*half)
                int kb = ((j & 4) << 2) + ((j & 3) << 1) + (half << 3);
                bf.u[j] = *(const u32*)&Bs[cur][nw + nt * 16 + lq][kb];
            }
            acc[nt] = __builtin_amdgcn_wmma_f32_16x16x32_bf16(
                          false, af.v, false, bf.v, (short)0, acc[nt], false, false);
        }
    }

    #pragma unroll
    for (int nt = 0; nt < 4; ++nt) {
        int n = n0 + nw + nt * 16 + lq;
        #pragma unroll
        for (int r = 0; r < 8; ++r) {
            int m = m0 + mw + r + (half << 3);
            float v = acc[nt][r] + (bias ? bias[m] : 0.0f);
            if (relu) v = fmaxf(v, 0.0f);
            out[(size_t)n * M + m] = f2bf(v);
        }
    }
}

// ---------------------------------------------------------------------------
// Spiral conv as fused-gather WMMA GEMM.
//   out[n*M+m] = relu( sum_kk W[m,kk] * zin[b, idx[v, kk/C], kk%C] + bias[m] )
// n = b*Nf + v, K = SP*C.  C%32==0 -> each 32-wide k-tile lies in one spiral
// tap s, so the gathered B tile is one contiguous 16B async copy per column.
// Double-buffered like k_gemm. N%128==0, K%32==0; M arbitrary (dead A rows
// zero-filled in LDS once; their outputs never stored).
// ---------------------------------------------------------------------------
__global__ __launch_bounds__(256)
void k_gemm_spiral(const u16* __restrict__ A, const u16* __restrict__ zin,
                   const int* __restrict__ idx, u16* __restrict__ out,
                   const float* __restrict__ bias,
                   int M, int N, int K, int C, int Nf, int SP)
{
    __shared__ u16 As[2][64][40];
    __shared__ u16 Bs[2][128][40];

    const int t    = threadIdx.x;
    const int m0   = blockIdx.y * 64;
    const int n0   = blockIdx.x * 128;
    const int wave = t >> 5, lane = t & 31;
    const int lq   = lane & 15, half = lane >> 4;
    const int mw   = (wave >> 1) << 4;
    const int nw   = (wave & 1) << 6;

    v8f acc[4] = {v8f{}, v8f{}, v8f{}, v8f{}};

    const int a_m = t >> 2, a_k = (t & 3) << 3;
    const int g_n = t >> 2, g_k = (t & 3) << 3;

    const bool a_ok = (m0 + a_m) < M;
    if (!a_ok) {                        // zero-fill dead A rows (both buffers)
        uint4 z = make_uint4(0u, 0u, 0u, 0u);
        *(uint4*)&As[0][a_m][a_k] = z;
        *(uint4*)&As[1][a_m][a_k] = z;
    }

    const u32 dA[2]  = {lds_off(&As[0][a_m][a_k]),      lds_off(&As[1][a_m][a_k])};
    const u32 dB0[2] = {lds_off(&Bs[0][g_n][g_k]),      lds_off(&Bs[1][g_n][g_k])};
    const u32 dB1[2] = {lds_off(&Bs[0][g_n + 64][g_k]), lds_off(&Bs[1][g_n + 64][g_k])};

    const int n1 = n0 + g_n,      b1 = n1 / Nf, v1 = n1 - b1 * Nf;
    const int n2 = n0 + 64 + g_n, b2 = n2 / Nf, v2 = n2 - b2 * Nf;

    // prologue: tile 0 (k0 = 0 -> tap s = 0, channel = g_k)
    if (a_ok)
        async_ld_b128(dA[0], A + (size_t)(m0 + a_m) * K + a_k);
    {
        const int r1 = idx[v1 * SP];
        const int r2 = idx[v2 * SP];
        async_ld_b128(dB0[0], zin + ((size_t)(b1 * Nf + r1) * C + g_k));
        async_ld_b128(dB1[0], zin + ((size_t)(b2 * Nf + r2) * C + g_k));
    }

    const int nk = K >> 5;
    for (int i = 0; i < nk; ++i) {
        const int cur = i & 1, nxt = cur ^ 1;

        wait_async0();
        __syncthreads();

        if (i + 1 < nk) {               // prefetch tile i+1 (gathered)
            const int k1 = (i + 1) << 5;
            if (a_ok)
                async_ld_b128(dA[nxt], A + (size_t)(m0 + a_m) * K + (k1 + a_k));
            const int s  = k1 / C;                 // spiral tap of next k-tile
            const int c0 = (k1 - s * C) + g_k;     // channel offset within tap
            const int r1 = idx[v1 * SP + s];
            const int r2 = idx[v2 * SP + s];
            async_ld_b128(dB0[nxt], zin + ((size_t)(b1 * Nf + r1) * C + c0));
            async_ld_b128(dB1[nxt], zin + ((size_t)(b2 * Nf + r2) * C + c0));
        }

        Frag16 af;
        #pragma unroll
        for (int j = 0; j < 8; ++j) {
            int ka = ((j & 4) << 2) + ((j & 3) << 1) + (half << 3);
            af.u[j] = *(const u32*)&As[cur][mw + lq][ka];
        }
        #pragma unroll
        for (int nt = 0; nt < 4; ++nt) {
            Frag16 bf;
            #pragma unroll
            for (int j = 0; j < 8; ++j) {
                int kb = ((j & 4) << 2) + ((j & 3) << 1) + (half << 3);
                bf.u[j] = *(const u32*)&Bs[cur][nw + nt * 16 + lq][kb];
            }
            acc[nt] = __builtin_amdgcn_wmma_f32_16x16x32_bf16(
                          false, af.v, false, bf.v, (short)0, acc[nt], false, false);
        }
    }

    #pragma unroll
    for (int nt = 0; nt < 4; ++nt) {
        int n = n0 + nw + nt * 16 + lq;
        #pragma unroll
        for (int r = 0; r < 8; ++r) {
            int m = m0 + mw + r + (half << 3);
            if (m >= M) continue;
            float v = fmaxf(acc[nt][r] + (bias ? bias[m] : 0.0f), 0.0f);
            out[(size_t)n * M + m] = f2bf(v);
        }
    }
}

// --------------------------- small helper kernels --------------------------

__global__ void k_f32_to_bf16(const float* __restrict__ src, u16* __restrict__ dst, int n)
{
    for (int i = blockIdx.x * blockDim.x + threadIdx.x; i < n; i += gridDim.x * blockDim.x)
        dst[i] = f2bf(src[i]);
}

// Xc column-major [K=1024 x N=2048]: Xc[(b*16+l)*1024 + c] = j[b,c,l] + r[b,c,l]
__global__ void k_pack_add_jr(const float* __restrict__ j, const float* __restrict__ r,
                              u16* __restrict__ dst)
{
    int id = blockIdx.x * blockDim.x + threadIdx.x;
    if (id >= 2048 * 1024) return;
    int n = id >> 10, c = id & 1023;
    int b = n >> 4, l = n & 15;
    size_t si = (size_t)b * 16384 + (size_t)c * 16 + l;
    dst[id] = f2bf(j[si] + r[si]);
}

// Xf column-major [K=256 x N=8192]: Xf[(b*64+p)*256 + c] = x[b,c,p]
__global__ void k_pack_x(const float* __restrict__ x, u16* __restrict__ dst)
{
    int id = blockIdx.x * blockDim.x + threadIdx.x;
    if (id >= 8192 * 256) return;
    int n = id >> 8, c = id & 255;
    int b = n >> 6, p = n & 63;
    dst[id] = f2bf(x[(size_t)b * 16384 + (size_t)c * 64 + p]);
}

// Z column-major [K=4096 x N=128]: Z[b*4096 + c*16 + l] = Y2[(b*16+l)*256 + c]
__global__ void k_repack_z(const u16* __restrict__ y2, u16* __restrict__ z)
{
    int id = blockIdx.x * blockDim.x + threadIdx.x;
    if (id >= 128 * 4096) return;
    int b = id >> 12, k = id & 4095;
    int c = k >> 4, l = k & 15;
    z[id] = y2[(size_t)(b * 16 + l) * 256 + c];
}

// Bilinear grid sample: feat col-major [(b*64+pix)*256 + c] -> zgs f32 [B,21,256]
__global__ void k_grid_sample(const u16* __restrict__ featb,
                              const float* __restrict__ uv, float* __restrict__ zgs)
{
    int bk = blockIdx.x;               // b*21 + k
    int b  = bk / 21;
    int c  = threadIdx.x;              // 256 channels
    float u  = uv[(size_t)bk * 2 + 0];
    float vv = uv[(size_t)bk * 2 + 1];
    float un = fminf(fmaxf((u  - 0.5f) * 2.0f, -1.0f), 1.0f);
    float vn = fminf(fmaxf((vv - 0.5f) * 2.0f, -1.0f), 1.0f);
    float gx = (un + 1.0f) * 0.5f * 7.0f;
    float gy = (vn + 1.0f) * 0.5f * 7.0f;
    float x0f = floorf(gx), y0f = floorf(gy);
    float wx = gx - x0f, wy = gy - y0f;
    int x0 = min(max((int)x0f, 0), 7);
    int x1 = min(x0 + 1, 7);
    int y0 = min(max((int)y0f, 0), 7);
    int y1 = min(y0 + 1, 7);
    size_t base = (size_t)b * 64 * 256 + c;
    float f00 = bf2f(featb[base + (y0 * 8 + x0) * 256]);
    float f01 = bf2f(featb[base + (y0 * 8 + x1) * 256]);
    float f10 = bf2f(featb[base + (y1 * 8 + x0) * 256]);
    float f11 = bf2f(featb[base + (y1 * 8 + x1) * 256]);
    zgs[(size_t)bk * 256 + c] = f00 * (1.0f - wx) * (1.0f - wy)
                              + f01 * wx * (1.0f - wy)
                              + f10 * (1.0f - wx) * wy
                              + f11 * wx * wy;
}

// zup[(b*49+v)*256 + c] = bf16( sum_u up[v,u] * zgs[b,u,c] )
__global__ void k_upsample(const float* __restrict__ zgs, const float* __restrict__ up,
                           u16* __restrict__ zup)
{
    int id = blockIdx.x * blockDim.x + threadIdx.x;
    if (id >= 128 * 49 * 256) return;
    int c = id & 255;
    int rest = id >> 8;
    int v = rest % 49, b = rest / 49;
    float acc = 0.0f;
    #pragma unroll
    for (int u = 0; u < 21; ++u)
        acc += up[v * 21 + u] * zgs[((size_t)b * 21 + u) * 256 + c];
    zup[id] = f2bf(acc);
}

// Dense 3-tap pool-up (row = repeat(arange(nf),3) -> no atomics needed)
__global__ void k_pool(const u16* __restrict__ zin, const int* __restrict__ col,
                       const float* __restrict__ val, u16* __restrict__ out,
                       int Nc, int Nf, int C, int total)
{
    int id = blockIdx.x * blockDim.x + threadIdx.x;
    if (id >= total) return;
    int c = id % C;
    int rest = id / C;
    int v = rest % Nf, b = rest / Nf;
    float acc = 0.0f;
    #pragma unroll
    for (int tt = 0; tt < 3; ++tt) {
        int i = 3 * v + tt;
        acc += val[i] * bf2f(zin[((size_t)b * Nc + col[i]) * C + c]);
    }
    out[id] = f2bf(acc);
}

// Head: pred[b,v,o] = sum_{s,c} z[b, idx[v,s], c] * Wh[o, s*32+c] + bh[o]
__global__ void k_head(const u16* __restrict__ z, const int* __restrict__ idx,
                       const float* __restrict__ Wh, const float* __restrict__ bh,
                       float* __restrict__ out)
{
    int id = blockIdx.x * blockDim.x + threadIdx.x;   // b*778 + v
    if (id >= 128 * 778) return;
    int v = id % 778, b = id / 778;
    float a0 = bh[0], a1 = bh[1], a2 = bh[2];
    for (int s = 0; s < 9; ++s) {
        int r = idx[v * 9 + s];
        const u16* zp = z + ((size_t)b * 778 + r) * 32;
        #pragma unroll
        for (int c = 0; c < 32; ++c) {
            float zv = bf2f(zp[c]);
            int wk = s * 32 + c;
            a0 += zv * Wh[0 * 288 + wk];
            a1 += zv * Wh[1 * 288 + wk];
            a2 += zv * Wh[2 * 288 + wk];
        }
    }
    float* o = out + (size_t)id * 3;
    o[0] = a0; o[1] = a1; o[2] = a2;
}

// pred_rot[b,o] = sum_k F3[b*256 + k] * fW4[o,k] + fb4[o]   (M=6: plain VALU)
__global__ void k_rot_final(const u16* __restrict__ F3, const float* __restrict__ W,
                            const float* __restrict__ bias, float* __restrict__ out)
{
    int id = blockIdx.x * blockDim.x + threadIdx.x;   // b*6 + o
    if (id >= 128 * 6) return;
    int o = id % 6, b = id / 6;
    float acc = bias[o];
    for (int k = 0; k < 256; ++k)
        acc += bf2f(F3[(size_t)b * 256 + k]) * W[o * 256 + k];
    out[id] = acc;
}

// ---------------------------------------------------------------------------

extern "C" void kernel_launch(void* const* d_in, const int* in_sizes, int n_in,
                              void* d_out, int out_size, void* d_ws, size_t ws_size,
                              hipStream_t stream)
{
    (void)in_sizes; (void)n_in; (void)out_size; (void)ws_size;

    // ---- inputs (setup_inputs dict insertion order, nested in natural order)
    const float* uv   = (const float*)d_in[0];
    const float* x    = (const float*)d_in[1];
    const float* jmid = (const float*)d_in[2];
    const float* rmid = (const float*)d_in[3];
    const float* deW  = (const float*)d_in[4];
    const float* spW[4] = {(const float*)d_in[5], (const float*)d_in[7],
                           (const float*)d_in[9], (const float*)d_in[11]};
    const float* spB[4] = {(const float*)d_in[6], (const float*)d_in[8],
                           (const float*)d_in[10], (const float*)d_in[12]};
    const float* Wh  = (const float*)d_in[13];
    const float* bh  = (const float*)d_in[14];
    const float* up  = (const float*)d_in[15];
    const float* cW1 = (const float*)d_in[16]; const float* cb1 = (const float*)d_in[17];
    const float* cW2 = (const float*)d_in[18]; const float* cb2 = (const float*)d_in[19];
    const float* pW1 = (const float*)d_in[20]; const float* pb1 = (const float*)d_in[21];
    const float* pW2 = (const float*)d_in[22]; const float* pb2 = (const float*)d_in[23];
    const float* fW1 = (const float*)d_in[24]; const float* fb1 = (const float*)d_in[25];
    const float* fW2 = (const float*)d_in[26]; const float* fb2 = (const float*)d_in[27];
    const float* fW3 = (const float*)d_in[28]; const float* fb3 = (const float*)d_in[29];
    const float* fW4 = (const float*)d_in[30]; const float* fb4 = (const float*)d_in[31];
    const int* idxs[4] = {(const int*)d_in[32], (const int*)d_in[33],
                          (const int*)d_in[34], (const int*)d_in[35]};
    const int*   pcol[4] = {(const int*)d_in[37], (const int*)d_in[40],
                            (const int*)d_in[43], (const int*)d_in[46]};
    const float* pval[4] = {(const float*)d_in[38], (const float*)d_in[41],
                            (const float*)d_in[44], (const float*)d_in[47]};

    float* out_pred = (float*)d_out;
    float* out_rot  = (float*)d_out + (size_t)128 * 778 * 3;

    // ---- workspace arena
    char* ws = (char*)d_ws;
    size_t off = 0;
    auto alloc = [&](size_t bytes) -> char* {
        size_t p = (off + 255) & ~(size_t)255;
        off = p + bytes;
        return ws + p;
    };

    // bf16 weight copies
    u16* cW1b = (u16*)alloc(512 * 1024 * 2);
    u16* cW2b = (u16*)alloc(256 * 512 * 2);
    u16* pW1b = (u16*)alloc((size_t)2048 * 4096 * 2);
    u16* pW2b = (u16*)alloc((size_t)1024 * 2048 * 2);
    u16* fW1b = (u16*)alloc((size_t)1024 * 1024 * 2);
    u16* fW2b = (u16*)alloc(512 * 1024 * 2);
    u16* fW3b = (u16*)alloc(256 * 512 * 2);
    u16* deWb = (u16*)alloc(256 * 256 * 2);
    u16* spWb[4];
    spWb[0] = (u16*)alloc(256 * 2304 * 2);
    spWb[1] = (u16*)alloc(128 * 2304 * 2);
    spWb[2] = (u16*)alloc(64 * 1152 * 2);
    spWb[3] = (u16*)alloc(32 * 576 * 2);

    // activations (all column-major)
    u16* Xc  = (u16*)alloc((size_t)2048 * 1024 * 2);
    u16* Y1  = (u16*)alloc((size_t)2048 * 512 * 2);
    u16* Y2  = (u16*)alloc((size_t)2048 * 256 * 2);
    u16* Zm  = (u16*)alloc((size_t)128 * 4096 * 2);
    u16* P1  = (u16*)alloc((size_t)128 * 2048 * 2);
    u16* P2  = (u16*)alloc((size_t)128 * 1024 * 2);
    u16* F1  = (u16*)alloc((size_t)128 * 1024 * 2);
    u16* F2  = (u16*)alloc((size_t)128 * 512 * 2);
    u16* F3  = (u16*)alloc((size_t)128 * 256 * 2);
    u16* Xf  = (u16*)alloc((size_t)8192 * 256 * 2);
    u16* Ft  = (u16*)alloc((size_t)8192 * 256 * 2);
    float* Zg = (float*)alloc((size_t)128 * 21 * 256 * 4);
    u16* Zu  = (u16*)alloc((size_t)128 * 49 * 256 * 2);
    u16* pl0 = (u16*)alloc((size_t)128 * 98 * 256 * 2);
    u16* a0  = (u16*)alloc((size_t)128 * 98 * 256 * 2);
    u16* pl1 = (u16*)alloc((size_t)128 * 195 * 256 * 2);
    u16* a1  = (u16*)alloc((size_t)128 * 195 * 128 * 2);
    u16* pl2 = (u16*)alloc((size_t)128 * 389 * 128 * 2);
    u16* a2  = (u16*)alloc((size_t)128 * 389 * 64 * 2);
    u16* pl3 = (u16*)alloc((size_t)128 * 778 * 64 * 2);
    u16* a3  = (u16*)alloc((size_t)128 * 778 * 32 * 2);

    auto cvt = [&](const float* s, u16* d, int n) {
        hipLaunchKernelGGL(k_f32_to_bf16, dim3((n + 255) / 256), dim3(256), 0, stream, s, d, n);
    };
    auto gemm = [&](const u16* A, const u16* B, u16* O, const float* bias,
                    int M, int N, int K, int relu) {
        dim3 g(N / 128, M / 64);
        hipLaunchKernelGGL(k_gemm, g, dim3(256), 0, stream, A, B, O, bias, M, N, K, relu);
    };
    auto sgemm = [&](const u16* A, const u16* zin, const int* idx, u16* O, const float* bias,
                     int M, int Nf, int K, int C) {
        int N = 128 * Nf;
        dim3 g(N / 128, (M + 63) / 64);
        hipLaunchKernelGGL(k_gemm_spiral, g, dim3(256), 0, stream,
                           A, zin, idx, O, bias, M, N, K, C, Nf, 9);
    };

    // ---- weight conversions (independent)
    cvt(cW1, cW1b, 512 * 1024);
    cvt(cW2, cW2b, 256 * 512);
    cvt(pW1, pW1b, 2048 * 4096);
    cvt(pW2, pW2b, 1024 * 2048);
    cvt(fW1, fW1b, 1024 * 1024);
    cvt(fW2, fW2b, 512 * 1024);
    cvt(fW3, fW3b, 256 * 512);
    cvt(deW, deWb, 256 * 256);
    cvt(spW[0], spWb[0], 256 * 2304);
    cvt(spW[1], spWb[1], 128 * 2304);
    cvt(spW[2], spWb[2], 64 * 1152);
    cvt(spW[3], spWb[3], 32 * 576);

    // ---- rot_reg branch
    hipLaunchKernelGGL(k_pack_add_jr, dim3((2048 * 1024 + 255) / 256), dim3(256), 0, stream,
                       jmid, rmid, Xc);
    gemm(cW1b, Xc, Y1, cb1, 512, 2048, 1024, 1);
    gemm(cW2b, Y1, Y2, cb2, 256, 2048, 512, 1);
    hipLaunchKernelGGL(k_repack_z, dim3((128 * 4096 + 255) / 256), dim3(256), 0, stream, Y2, Zm);
    gemm(pW1b, Zm, P1, pb1, 2048, 128, 4096, 1);
    gemm(pW2b, P1, P2, pb2, 1024, 128, 2048, 1);
    gemm(fW1b, P2, F1, fb1, 1024, 128, 1024, 1);
    gemm(fW2b, F1, F2, fb2, 512, 128, 1024, 1);
    gemm(fW3b, F2, F3, fb3, 256, 128, 512, 1);
    hipLaunchKernelGGL(k_rot_final, dim3(3), dim3(256), 0, stream, F3, fW4, fb4, out_rot);

    // ---- mesh branch: 1x1 conv -> grid sample -> upsample
    hipLaunchKernelGGL(k_pack_x, dim3((8192 * 256 + 255) / 256), dim3(256), 0, stream, x, Xf);
    gemm(deWb, Xf, Ft, nullptr, 256, 8192, 256, 0);
    hipLaunchKernelGGL(k_grid_sample, dim3(128 * 21), dim3(256), 0, stream, Ft, uv, Zg);
    hipLaunchKernelGGL(k_upsample, dim3((128 * 49 * 256 + 255) / 256), dim3(256), 0, stream,
                       Zg, up, Zu);

    // ---- spiral decoder: coarse -> fine  (layer i uses pools[3-i], indices[3-i])
    int tot;
    tot = 128 * 98 * 256;
    hipLaunchKernelGGL(k_pool, dim3((tot + 255) / 256), dim3(256), 0, stream,
                       Zu, pcol[3], pval[3], pl0, 49, 98, 256, tot);
    sgemm(spWb[0], pl0, idxs[3], a0, spB[0], 256, 98, 2304, 256);

    tot = 128 * 195 * 256;
    hipLaunchKernelGGL(k_pool, dim3((tot + 255) / 256), dim3(256), 0, stream,
                       a0, pcol[2], pval[2], pl1, 98, 195, 256, tot);
    sgemm(spWb[1], pl1, idxs[2], a1, spB[1], 128, 195, 2304, 256);

    tot = 128 * 389 * 128;
    hipLaunchKernelGGL(k_pool, dim3((tot + 255) / 256), dim3(256), 0, stream,
                       a1, pcol[1], pval[1], pl2, 195, 389, 128, tot);
    sgemm(spWb[2], pl2, idxs[1], a2, spB[2], 64, 389, 1152, 128);

    tot = 128 * 778 * 64;
    hipLaunchKernelGGL(k_pool, dim3((tot + 255) / 256), dim3(256), 0, stream,
                       a2, pcol[0], pval[0], pl3, 389, 778, 64, tot);
    sgemm(spWb[3], pl3, idxs[0], a3, spB[3], 32, 778, 576, 64);

    // ---- head -> pred [B,778,3]
    hipLaunchKernelGGL(k_head, dim3((128 * 778 + 255) / 256), dim3(256), 0, stream,
                       a3, idxs[0], Wh, bh, out_pred);
}
